// Attention_46712064311988
// MI455X (gfx1250) — matrix operand order repaired and checked by reading, not compiled, not run on hardware
//
#include <hip/hip_runtime.h>
#include <hip/hip_bf16.h>
#include <stdint.h>

typedef __bf16 bf16;
typedef __attribute__((ext_vector_type(16))) __bf16 v16bf;
typedef __attribute__((ext_vector_type(8)))  float  v8f;
typedef __attribute__((ext_vector_type(4)))  unsigned int u32x4;
typedef __attribute__((ext_vector_type(8)))  int i32x8;
typedef __attribute__((ext_vector_type(4)))  int i32x4;

union FragA { v16bf v; uint32_t u[8]; };

#define WMMA_BF16(A_, B_, C_) \
  __builtin_amdgcn_wmma_f32_16x16x32_bf16(false, (A_), false, (B_), (short)0, (C_), false, false)

#if __has_builtin(__builtin_amdgcn_tensor_load_to_lds) && \
    __has_builtin(__builtin_amdgcn_s_wait_tensorcnt)
#define USE_TDM 1
#else
#define USE_TDM 0
#endif

// LDS byte offset of a __shared__ object: flat LDS addresses carry the
// aperture in the high 32 bits and the LDS offset in the low 32 bits.
#define LDS_OFF(p) ((uint32_t)(uintptr_t)(void*)(p))

// Wave-leader test as a *scalar* (SGPR) condition: tensor ops ignore EXEC,
// so the non-leader waves must take a scalar branch around the TDM issue,
// not an EXEC-masked fallthrough.
#define IS_WAVE0(tid) (__builtin_amdgcn_readfirstlane((tid) >> 5) == 0)

#if USE_TDM
// 2-D TDM load: tile_d1 rows of tile_d0 bf16 elements, row stride stride0
// (elements), from gaddr into LDS at lds_off (packed row-major).
// D# bit layout per CDNA5 ISA ch.8: group0 = {count/type, lds_addr,
// global_addr}, group1 = {data_size, tensor dims, tile dims, stride}.
// Groups 2/3 and the trailing group are 3D-5D / gather fields: zero here.
__device__ __forceinline__ void tdm_load_2d(uint32_t lds_off, const void* gaddr,
                                            uint32_t tensor_d0, uint32_t tensor_d1,
                                            uint32_t stride0,
                                            uint32_t tile_d0, uint32_t tile_d1) {
  const uint64_t ga = (uint64_t)(uintptr_t)gaddr;
  u32x4 g0;
  g0[0] = 1u;                                   // count=1, user mode
  g0[1] = lds_off;                              // lds_addr
  g0[2] = (uint32_t)ga;                         // global_addr[31:0]
  g0[3] = (uint32_t)(ga >> 32) | (2u << 30);    // global_addr[56:32] | type=2
  i32x8 g1;
  g1[0] = (int)(1u << 16);                      // data_size=1 -> 2 bytes
  g1[1] = (int)((tensor_d0 & 0xffffu) << 16);                       // dim0 lo
  g1[2] = (int)((tensor_d0 >> 16) | ((tensor_d1 & 0xffffu) << 16)); // dim0 hi|dim1 lo
  g1[3] = (int)((tensor_d1 >> 16) | (tile_d0 << 16));               // dim1 hi|tile0
  g1[4] = (int)(tile_d1 & 0xffffu);             // tile_dim1 | tile_dim2=0
  g1[5] = (int)stride0;                         // tensor_dim0_stride[31:0]
  g1[6] = 0;
  g1[7] = 0;
  const i32x4 z4 = {0, 0, 0, 0};
  const i32x8 z8 = {0, 0, 0, 0, 0, 0, 0, 0};
  __builtin_amdgcn_tensor_load_to_lds(g0, g1, z4, z4, z8, 0);
}
#endif

// ---------------------------------------------------------------------------
// Stage 1: fp32 -> bf16 conversion (float4 in, 4xbf16 (8B) out)
// ---------------------------------------------------------------------------
__global__ void __launch_bounds__(256, 4)
cvt_f32_bf16_v4(const float* __restrict__ src, bf16* __restrict__ dst, int n4) {
  int i = blockIdx.x * blockDim.x + threadIdx.x;
  if (i >= n4) return;
  float4 f = ((const float4*)src)[i];
  bf16 h[4];
  h[0] = (bf16)f.x; h[1] = (bf16)f.y; h[2] = (bf16)f.z; h[3] = (bf16)f.w;
  ((uint2*)dst)[i] = *(const uint2*)h;
}

// ---------------------------------------------------------------------------
// Stage 2/4: tiled bf16 GEMM  out[M,D] = A[M,K] * W[D,K]^T   (K = 1024)
// 64x64 tile per 256-thread block (8 waves). Double-buffered LDS tiles
// filled by the Tensor Data Mover (wave 0 issues, all waves compute).
// MODE 0: QKV epilogue (split q/k/v per head, scale q, transpose v)
// MODE 1: proj epilogue (add bias, fp32 out)
// ---------------------------------------------------------------------------
template <int MODE>
__global__ void __launch_bounds__(256, 1)
gemm_bf16_64x64(const bf16* __restrict__ A,
                const bf16* __restrict__ W,
                bf16* __restrict__ qb, bf16* __restrict__ kb,
                bf16* __restrict__ vbT,
                float* __restrict__ outp,
                const float* __restrict__ bias) {
  constexpr int K = 1024;
  __shared__ __align__(16) bf16 At[2][64 * 32];
  __shared__ __align__(16) bf16 Bt[2][64 * 32];

  const int t    = threadIdx.x;      // 0..255
  const int wave = t >> 5;
  const int lane = t & 31;
  const int half = lane >> 4;
  const int l16  = lane & 15;
  const int row0 = blockIdx.y * 64;
  const int col0 = blockIdx.x * 64;
  const int wr   = (wave & 3) * 16;  // wave row offset in tile
  const int wc   = (wave >> 2) * 32; // wave col offset in tile

  const bf16* gA = A + (size_t)row0 * K;
  const bf16* gB = W + (size_t)col0 * K;

  v8f zero = {};
  v8f acc0 = zero, acc1 = zero;

#if USE_TDM
  const uint32_t atOff[2] = {LDS_OFF(At[0]), LDS_OFF(At[1])};
  const uint32_t btOff[2] = {LDS_OFF(Bt[0]), LDS_OFF(Bt[1])};
  if (IS_WAVE0(t)) {  // preload k-chunk 0 into buffer 0
    tdm_load_2d(atOff[0], gA, K, 64, K, 32, 64);
    tdm_load_2d(btOff[0], gB, K, 64, K, 32, 64);
  }
#else
  const int sr = t >> 2;            // staging row 0..63
  const int sc = (t & 3) * 8;       // staging col offset (bf16 elems)
#endif

  for (int it = 0; it < K / 32; ++it) {
    const int buf = it & 1;
#if USE_TDM
    if (IS_WAVE0(t)) __builtin_amdgcn_s_wait_tensorcnt(0);  // current buf ready
    __syncthreads();  // also proves all waves finished reading buf^1
    if (it + 1 < K / 32 && IS_WAVE0(t)) {
      const int kn = (it + 1) * 32;
      tdm_load_2d(atOff[buf ^ 1], gA + kn, K, 64, K, 32, 64);
      tdm_load_2d(btOff[buf ^ 1], gB + kn, K, 64, K, 32, 64);
    }
#else
    const int k0 = it * 32;
    *(uint4*)(At[buf] + sr * 32 + sc) = *(const uint4*)(gA + (size_t)sr * K + k0 + sc);
    *(uint4*)(Bt[buf] + sr * 32 + sc) = *(const uint4*)(gB + (size_t)sr * K + k0 + sc);
    __syncthreads();
#endif

    FragA a, b0, b1;
#pragma unroll
    for (int p = 0; p < 8; ++p) {
      const int kA = 2 * p + ((p >= 4) ? 8 : 0) + 8 * half;  // A-frag K map
      const int kB = 2 * p + ((p >= 4) ? 8 : 0) + 8 * half;  // B-frag K map
      a.u[p]  = *(const uint32_t*)(At[buf] + (wr + l16) * 32 + kA);
      b0.u[p] = *(const uint32_t*)(Bt[buf] + (wc + l16) * 32 + kB);
      b1.u[p] = *(const uint32_t*)(Bt[buf] + (wc + 16 + l16) * 32 + kB);
    }
    acc0 = WMMA_BF16(a.v, b0.v, acc0);
    acc1 = WMMA_BF16(a.v, b1.v, acc1);
#if !USE_TDM
    __syncthreads();
#endif
  }

  // --- epilogue: C-frag layout: elem j -> row j + 8*half, col = base + l16 ---
#pragma unroll
  for (int j = 0; j < 8; ++j) {
    const int r = row0 + wr + j + 8 * half;   // global row in [0, 8192)
#pragma unroll
    for (int tc = 0; tc < 2; ++tc) {
      const int   d   = col0 + wc + 16 * tc + l16;
      const float val = (tc == 0) ? acc0[j] : acc1[j];
      if (MODE == 0) {
        const int which = d >> 10;            // 0=q 1=k 2=v
        const int c     = d & 1023;
        const int h     = c >> 6;
        const int hd    = c & 63;
        const int bI    = r >> 11;
        const int n     = r & 2047;
        const size_t hrow = (size_t)(bI * 16 + h) * 2048 + n;
        if (which == 0)      qb[hrow * 64 + hd] = (bf16)(val * 0.125f); // 1/sqrt(64)
        else if (which == 1) kb[hrow * 64 + hd] = (bf16)val;
        else vbT[((size_t)(bI * 16 + h) * 64 + hd) * 2048 + n] = (bf16)val;
      } else {
        outp[(size_t)r * 1024 + d] = val + bias[d];
      }
    }
  }
}

// ---------------------------------------------------------------------------
// Stage 3: flash attention. One (b,h) x 64 q-rows per 128-thread block
// (4 waves x 16 q rows). K/V tiles double-buffered in LDS, filled by TDM;
// 8 WMMAs for S = Q K^T, online softmax (batched wave32 half-group
// shuffles), P restaged through per-wave LDS as bf16 A-frags, 8 WMMAs for
// O += P V.  __launch_bounds__(128,1): few fat waves, no spills; latency
// hidden by the TDM prefetch pipeline, not by occupancy.
// ---------------------------------------------------------------------------
__global__ void __launch_bounds__(128, 1)
attn_flash_64(const bf16* __restrict__ qb,
              const bf16* __restrict__ kb,
              const bf16* __restrict__ vbT,
              bf16* __restrict__ ob) {
  __shared__ __align__(16) bf16 Kl[2][64 * 64];   // [key][c]   2x8KB
  __shared__ __align__(16) bf16 Vl[2][64 * 64];   // [c][key]   2x8KB
  __shared__ __align__(16) bf16 Pl[4 * 16 * 64];  // per-wave P 8KB

  const int t    = threadIdx.x;  // 0..127
  const int wave = t >> 5;
  const int lane = t & 31;
  const int half = lane >> 4;
  const int l16  = lane & 15;
  const int bh   = blockIdx.y;               // b*16 + h
  const int q0   = blockIdx.x * 64 + wave * 16;
  const size_t baseRow = (size_t)bh * 2048;

  // --- load Q fragments (16 rows x 64 = two 16x32 A-frags), kept in VGPRs ---
  FragA qf[2];
  {
    const bf16* qrow = qb + (baseRow + q0 + l16) * 64;
#pragma unroll
    for (int ch = 0; ch < 2; ++ch)
#pragma unroll
      for (int p = 0; p < 8; ++p) {
        const int kA = 32 * ch + 2 * p + ((p >= 4) ? 8 : 0) + 8 * half;
        qf[ch].u[p] = *(const uint32_t*)(qrow + kA);
      }
  }

  v8f zero = {};
  v8f o[4] = {zero, zero, zero, zero};
  float m[8], l[8];
#pragma unroll
  for (int j = 0; j < 8; ++j) { m[j] = -1e30f; l[j] = 0.0f; }

  const bf16* gK = kb + baseRow * 64;             // [2048][64] contiguous
  const bf16* gV = vbT + (size_t)bh * 64 * 2048;  // [64][2048]
  bf16* Pw = Pl + wave * 16 * 64;

#if USE_TDM
  const uint32_t klOff[2] = {LDS_OFF(Kl[0]), LDS_OFF(Kl[1])};
  const uint32_t vlOff[2] = {LDS_OFF(Vl[0]), LDS_OFF(Vl[1])};
  if (IS_WAVE0(t)) {  // preload key-chunk 0 into buffer 0
    tdm_load_2d(klOff[0], gK, 64, 2048, 64, 64, 64);
    tdm_load_2d(vlOff[0], gV, 2048, 64, 2048, 64, 64);
  }
#endif

  for (int it = 0; it < 32; ++it) {
    const int buf = it & 1;
#if USE_TDM
    if (IS_WAVE0(t)) __builtin_amdgcn_s_wait_tensorcnt(0);  // current buf ready
    __syncthreads();  // all waves done reading buf^1 in the previous iter
    if (it + 1 < 32 && IS_WAVE0(t)) {
      const int kn = (it + 1) * 64;
      tdm_load_2d(klOff[buf ^ 1], gK + (size_t)kn * 64, 64, 2048, 64, 64, 64);
      tdm_load_2d(vlOff[buf ^ 1], gV + kn, 2048, 64, 2048, 64, 64);
    }
#else
    const int key0 = it * 64;
    {
      const uint4* sK = (const uint4*)(gK + (size_t)key0 * 64);
      uint4* dK = (uint4*)Kl[buf];
#pragma unroll
      for (int i = 0; i < 4; ++i) dK[t + 128 * i] = sK[t + 128 * i];
#pragma unroll
      for (int i = 0; i < 4; ++i) {
        const int flat = t + 128 * i;
        const int c = flat >> 3, jj = flat & 7;
        *(uint4*)(Vl[buf] + c * 64 + jj * 8) =
            *(const uint4*)(gV + (size_t)c * 2048 + key0 + jj * 8);
      }
    }
    __syncthreads();
#endif
    const bf16* Kc = Kl[buf];
    const bf16* Vc = Vl[buf];

    // --- S[16q x 64k] = Q (16x64) * K_chunk^T : 8 WMMAs ---
    v8f s[4] = {zero, zero, zero, zero};
#pragma unroll
    for (int tt = 0; tt < 4; ++tt)
#pragma unroll
      for (int ch = 0; ch < 2; ++ch) {
        FragA b;
#pragma unroll
        for (int p = 0; p < 8; ++p) {
          const int kB = 32 * ch + 2 * p + ((p >= 4) ? 8 : 0) + 8 * half;
          b.u[p] = *(const uint32_t*)(Kc + (16 * tt + l16) * 64 + kB);
        }
        s[tt] = WMMA_BF16(qf[ch].v, b.v, s[tt]);
      }

    // --- online softmax; row j+8*half lives across one 16-lane half group.
    //     Reductions batched step-outer so 8 bpermutes overlap per wait. ---
    float mx[8], sc[8], rs[8];
#pragma unroll
    for (int j = 0; j < 8; ++j)
      mx[j] = fmaxf(fmaxf(s[0][j], s[1][j]), fmaxf(s[2][j], s[3][j]));
#pragma unroll
    for (int off = 1; off < 16; off <<= 1)
#pragma unroll
      for (int j = 0; j < 8; ++j)
        mx[j] = fmaxf(mx[j], __shfl_xor(mx[j], off, 16));
#pragma unroll
    for (int j = 0; j < 8; ++j) {
      const float nm = fmaxf(m[j], mx[j]);
      sc[j] = __expf(m[j] - nm);
      m[j] = nm;
    }
#pragma unroll
    for (int j = 0; j < 8; ++j) {
      float r = 0.0f;
#pragma unroll
      for (int tt = 0; tt < 4; ++tt) {
        const float p0 = __expf(s[tt][j] - m[j]);
        s[tt][j] = p0;
        r += p0;
      }
      rs[j] = r;
    }
#pragma unroll
    for (int off = 1; off < 16; off <<= 1)
#pragma unroll
      for (int j = 0; j < 8; ++j)
        rs[j] += __shfl_xor(rs[j], off, 16);
#pragma unroll
    for (int j = 0; j < 8; ++j) l[j] = l[j] * sc[j] + rs[j];
#pragma unroll
    for (int ct = 0; ct < 4; ++ct)
#pragma unroll
      for (int j = 0; j < 8; ++j) o[ct][j] *= sc[j];

    // --- re-layout P: C-frag -> bf16 [16][64] in per-wave LDS region ---
#pragma unroll
    for (int tt = 0; tt < 4; ++tt)
#pragma unroll
      for (int j = 0; j < 8; ++j)
        Pw[(j + 8 * half) * 64 + 16 * tt + l16] = (bf16)s[tt][j];
    // same-wave DS ops complete in order: safe to reload as A-frags below

    // --- O += P[16x64] * V[64x64] : 8 WMMAs ---
#pragma unroll
    for (int ch = 0; ch < 2; ++ch) {
      FragA a;
#pragma unroll
      for (int p = 0; p < 8; ++p) {
        const int kA = 32 * ch + 2 * p + ((p >= 4) ? 8 : 0) + 8 * half;
        a.u[p] = *(const uint32_t*)(Pw + l16 * 64 + kA);
      }
#pragma unroll
      for (int ct = 0; ct < 4; ++ct) {
        FragA b;
#pragma unroll
        for (int p = 0; p < 8; ++p) {
          const int kB = 32 * ch + 2 * p + ((p >= 4) ? 8 : 0) + 8 * half;
          b.u[p] = *(const uint32_t*)(Vc + (16 * ct + l16) * 64 + kB);
        }
        o[ct] = WMMA_BF16(a.v, b.v, o[ct]);
      }
    }
#if !USE_TDM
    __syncthreads();
#endif
  }

  // --- normalize and write attention output as bf16 [B,N,C] for proj GEMM ---
  const int bI = bh >> 4, h = bh & 15;
#pragma unroll
  for (int j = 0; j < 8; ++j) {
    const float inv = 1.0f / l[j];
    const int r = q0 + j + 8 * half;
    const size_t orow = ((size_t)bI * 2048 + r) * 1024 + h * 64;
#pragma unroll
    for (int ct = 0; ct < 4; ++ct)
      ob[orow + 16 * ct + l16] = (bf16)(o[ct][j] * inv);
  }
}

// ---------------------------------------------------------------------------
extern "C" void kernel_launch(void* const* d_in, const int* in_sizes, int n_in,
                              void* d_out, int out_size, void* d_ws, size_t ws_size,
                              hipStream_t stream) {
  const float* x     = (const float*)d_in[0];  // [4,2048,1024]
  const float* Wqkv  = (const float*)d_in[1];  // [3072,1024]
  const float* Wproj = (const float*)d_in[2];  // [1024,1024]
  const float* bproj = (const float*)d_in[3];  // [1024]

  const size_t nX  = 4ull * 2048 * 1024;       // 8,388,608
  const size_t nWq = 3072ull * 1024;           // 3,145,728
  const size_t nWp = 1024ull * 1024;           // 1,048,576

  char* ws = (char*)d_ws;
  size_t off = 0;
  auto carve = [&](size_t bytes) -> void* {
    void* p = ws + off;
    off += (bytes + 255) & ~(size_t)255;
    return p;
  };
  bf16* xb     = (bf16*)carve(nX * 2);
  bf16* wqkvb  = (bf16*)carve(nWq * 2);
  bf16* wprojb = (bf16*)carve(nWp * 2);
  bf16* qbuf   = (bf16*)carve(nX * 2);  // [B,H,N,64]
  bf16* kbuf   = (bf16*)carve(nX * 2);  // [B,H,N,64]
  bf16* vbT    = (bf16*)carve(nX * 2);  // [B,H,64,N]
  bf16* obuf   = (bf16*)carve(nX * 2);  // [B,N,C]

  // 1) downconvert
  cvt_f32_bf16_v4<<<(int)(nX / 4 / 256),  256, 0, stream>>>(x,     xb,     (int)(nX / 4));
  cvt_f32_bf16_v4<<<(int)(nWq / 4 / 256), 256, 0, stream>>>(Wqkv,  wqkvb,  (int)(nWq / 4));
  cvt_f32_bf16_v4<<<(int)(nWp / 4 / 256), 256, 0, stream>>>(Wproj, wprojb, (int)(nWp / 4));

  // 2) QKV GEMM: [8192,1024] x [3072,1024]^T, split per head
  gemm_bf16_64x64<0><<<dim3(48, 128), 256, 0, stream>>>(
      xb, wqkvb, qbuf, kbuf, vbT, nullptr, nullptr);

  // 3) flash attention: grid = (qblocks=32, B*H=64)
  attn_flash_64<<<dim3(32, 64), 128, 0, stream>>>(qbuf, kbuf, vbT, obuf);

  // 4) proj GEMM + bias -> fp32 d_out
  gemm_bf16_64x64<1><<<dim3(16, 128), 256, 0, stream>>>(
      obuf, wprojb, nullptr, nullptr, nullptr, (float*)d_out, bproj);
}
